// DualGNN_77309411328099
// MI455X (gfx1250) — matrix operand order repaired and checked by reading, not compiled, not run on hardware
//
#include <hip/hip_runtime.h>
#include <hip/hip_bf16.h>

typedef __attribute__((ext_vector_type(16))) _Float16 v16h;
typedef __attribute__((ext_vector_type(8)))  float    v8f;

#define NEG_SLOPE 0.01f
#define BN_EPS    1e-5f
#define NGRAPH    64
#define NC        8
#define GC        4

// ---------------------------------------------------------------- utilities

__global__ void zero_f32(float* __restrict__ p, size_t n) {
    size_t i = (size_t)blockIdx.x * blockDim.x + threadIdx.x;
    if (i < n) p[i] = 0.0f;
}

// out-degree from src, in-degree from dst (as float accumulators)
__global__ void degree_accum(const int* __restrict__ src, const int* __restrict__ dst,
                             float* __restrict__ outdeg, float* __restrict__ indeg, int E) {
    int e = blockIdx.x * blockDim.x + threadIdx.x;
    if (e >= E) return;
    unsafeAtomicAdd(&outdeg[src[e]], 1.0f);
    unsafeAtomicAdd(&indeg[dst[e]], 1.0f);
}

// d[i] = rsqrt(max(d[i],1))  in place
__global__ void inv_sqrt_deg(float* __restrict__ d, int n) {
    int i = blockIdx.x * blockDim.x + threadIdx.x;
    if (i < n) d[i] = rsqrtf(fmaxf(d[i], 1.0f));
}

// ------------------------------------------------- edge scatter (hot path)
// One wave per edge: 32 lanes x float4 = 128 cols. Gather row x[src] (scaled
// by src out-degree norm), scatter-add to agg[dst] with native f32 atomics.
// The edge index streams are perfectly sequential -> prefetch ahead.
__global__ void scatter_add_128(const float* __restrict__ x, const float* __restrict__ sOut,
                                const int* __restrict__ src, const int* __restrict__ dst,
                                float* __restrict__ agg, int E) {
    int gid  = blockIdx.x * blockDim.x + threadIdx.x;
    int e    = gid >> 5;
    int lane = gid & 31;
    if (e >= E) return;
    if (lane == 0 && e + 4096 < E) {
        __builtin_prefetch(src + e + 4096, 0, 0);
        __builtin_prefetch(dst + e + 4096, 0, 0);
    }
    int s = src[e], d = dst[e];
    float sc = sOut[s];
    float4 v = ((const float4*)(x + (size_t)s * 128))[lane];
    float* out = agg + (size_t)d * 128 + lane * 4;
    unsafeAtomicAdd(out + 0, v.x * sc);
    unsafeAtomicAdd(out + 1, v.y * sc);
    unsafeAtomicAdd(out + 2, v.z * sc);
    unsafeAtomicAdd(out + 3, v.w * sc);
}

// scalar column (nodetype) aggregation for the concat layer
__global__ void scatter_scalar(const float* __restrict__ val, const float* __restrict__ sOut,
                               const int* __restrict__ src, const int* __restrict__ dst,
                               float* __restrict__ aggC, int E) {
    int e = blockIdx.x * blockDim.x + threadIdx.x;
    if (e >= E) return;
    unsafeAtomicAdd(&aggC[dst[e]], val[src[e]] * sOut[src[e]]);
}

__global__ void scale_vec(float* __restrict__ v, const float* __restrict__ s, int n) {
    int i = blockIdx.x * blockDim.x + threadIdx.x;
    if (i < n) v[i] *= s[i];
}

// ------------------------------------------------------- WMMA 128x128 GEMM
// C[r,:] = lrelu?( (A[r,:]*srow[r]) @ W(128x128) + bias [+ extraCol[r]*extraW[:]] )
// Block: 256 threads = 8 waves; each wave owns a 16-row strip, all 128 cols.
// W is staged in LDS as f16, pre-swizzled into B-fragment order so a lane's
// 16 K-values for a (kb,nt) tile are one contiguous 32B chunk (2x ds_load_b128).
// NOTE: A and Cout are workspace buffers padded by >=128 rows, so boundary
// handling is done by index clamping on loads and *unguarded* stores into the
// padding -- no divergent guards anywhere in the kernel body.

__device__ inline v16h load_a_frag(const float* __restrict__ Ar, int kb, int koff, float sc) {
    // 16-bit A 16x32 layout: lanes 0-15 -> K = kb*32 + {0..7, 16..23}
    //                        lanes16-31 -> K = kb*32 + {8..15, 24..31}
    const float4* p0 = (const float4*)(Ar + kb * 32 + koff);
    const float4* p1 = (const float4*)(Ar + kb * 32 + koff + 16);
    float4 a = p0[0], b = p0[1], c = p1[0], d = p1[1];
    v16h f;
    f[0]  = (_Float16)(a.x * sc); f[1]  = (_Float16)(a.y * sc);
    f[2]  = (_Float16)(a.z * sc); f[3]  = (_Float16)(a.w * sc);
    f[4]  = (_Float16)(b.x * sc); f[5]  = (_Float16)(b.y * sc);
    f[6]  = (_Float16)(b.z * sc); f[7]  = (_Float16)(b.w * sc);
    f[8]  = (_Float16)(c.x * sc); f[9]  = (_Float16)(c.y * sc);
    f[10] = (_Float16)(c.z * sc); f[11] = (_Float16)(c.w * sc);
    f[12] = (_Float16)(d.x * sc); f[13] = (_Float16)(d.y * sc);
    f[14] = (_Float16)(d.z * sc); f[15] = (_Float16)(d.w * sc);
    return f;
}

__device__ inline float maybe_lrelu(float v, bool lrelu) {
    return (!lrelu || v >= 0.0f) ? v : NEG_SLOPE * v;
}

template<bool LRELU, bool EXTRA>
__global__ __launch_bounds__(256) void gemm128_wmma(
    const float* __restrict__ A, const float* __restrict__ W,
    const float* __restrict__ bias, const float* __restrict__ srow,
    const float* __restrict__ extraCol, const float* __restrict__ extraW,
    float* __restrict__ Cout, int nrows) {

    __shared__ __align__(32) _Float16 ldsB[16384];  // 128x128 f16, fragment-swizzled

    int tid = threadIdx.x;
    // Stage weights: W[k][n] -> fragment (kb = k/32, nt = n/16),
// lane = (n&15) + (((k>>3)&1) ? 16 : 0), element t = (k&7) | ((k>>1)&8).
    for (int idx = tid; idx < 16384; idx += 256) {
        int k = idx >> 7, n = idx & 127;
        int kb = k >> 5, t = (k & 7) | ((k >> 1) & 8), hi = (k >> 3) & 1;
        int lane = (n & 15) + (hi ? 16 : 0);
        int nt = n >> 4;
        ldsB[((kb * 8 + nt) * 32 + lane) * 16 + t] = (_Float16)W[idx];
    }
    __syncthreads();

    int wave = tid >> 5, lane = tid & 31;
    int r0 = blockIdx.x * 128 + wave * 16;
    if (r0 >= nrows) return;   // fully-empty waves of the last block

    // clamp to a real row: padding-row results are stored into buffer padding
    int arow = min(r0 + (lane & 15), nrows - 1);
    const float* Ar = A + (size_t)arow * 128;
    float sc = srow[arow];
    int koff = (lane < 16) ? 0 : 8;

    v16h a0 = load_a_frag(Ar, 0, koff, sc);
    v16h a1 = load_a_frag(Ar, 1, koff, sc);
    v16h a2 = load_a_frag(Ar, 2, koff, sc);
    v16h a3 = load_a_frag(Ar, 3, koff, sc);

    int rofs = (lane < 16) ? 0 : 8;
    float ev[8];
    if (EXTRA) {
#pragma unroll
        for (int j = 0; j < 8; ++j)
            ev[j] = extraCol[min(r0 + j + rofs, nrows - 1)];
    }

    int colBase = lane & 15;
    for (int nt = 0; nt < 8; ++nt) {
        int col = nt * 16 + colBase;
        float bv = bias[col];
        v8f acc;
        if (EXTRA) {
            float ew = extraW[col];
#pragma unroll
            for (int j = 0; j < 8; ++j) acc[j] = bv + ev[j] * ew;
        } else {
#pragma unroll
            for (int j = 0; j < 8; ++j) acc[j] = bv;
        }

        const _Float16* bp = ldsB + (size_t)(nt * 32 + lane) * 16;
        v16h b0 = *(const v16h*)(bp);
        v16h b1 = *(const v16h*)(bp + 4096);
        v16h b2 = *(const v16h*)(bp + 8192);
        v16h b3 = *(const v16h*)(bp + 12288);

        acc = __builtin_amdgcn_wmma_f32_16x16x32_f16(false, a0, false, b0, (short)0, acc, false, false);
        acc = __builtin_amdgcn_wmma_f32_16x16x32_f16(false, a1, false, b1, (short)0, acc, false, false);
        acc = __builtin_amdgcn_wmma_f32_16x16x32_f16(false, a2, false, b2, (short)0, acc, false, false);
        acc = __builtin_amdgcn_wmma_f32_16x16x32_f16(false, a3, false, b3, (short)0, acc, false, false);

        // straight-line stores; tail rows land in buffer padding
        float* outBase = Cout + (size_t)(r0 + rofs) * 128 + col;
#pragma unroll
        for (int j = 0; j < 8; ++j)
            outBase[(size_t)j * 128] = maybe_lrelu(acc[j], LRELU);
    }
}

// ----------------------------------------------------------- batch norm
__global__ void col_stats(const float* __restrict__ X, float* __restrict__ csum,
                          float* __restrict__ csq, int nrows) {
    int c = threadIdx.x;  // blockDim.x == 128
    int r0 = blockIdx.x * 256;
    int r1 = r0 + 256; if (r1 > nrows) r1 = nrows;
    float s = 0.0f, q = 0.0f;
    for (int r = r0; r < r1; ++r) {
        float v = X[(size_t)r * 128 + c];
        s += v; q += v * v;
    }
    unsafeAtomicAdd(&csum[c], s);
    unsafeAtomicAdd(&csq[c], q);
}

__global__ void bn_finalize(const float* __restrict__ csum, const float* __restrict__ csq,
                            const float* __restrict__ gamma, const float* __restrict__ beta,
                            float* __restrict__ scale, float* __restrict__ shift, int nrows) {
    int c = threadIdx.x;  // 128
    float inv_n = 1.0f / (float)nrows;
    float mu = csum[c] * inv_n;
    float var = fmaxf(csq[c] * inv_n - mu * mu, 0.0f);
    float is = rsqrtf(var + BN_EPS) * gamma[c];
    scale[c] = is;
    shift[c] = beta[c] - mu * is;
}

// float4-wide: rows are 128 wide so (i4*4)&127 gives the column of lane's quad
__global__ void bn_apply_lrelu(float* __restrict__ X, const float* __restrict__ scale,
                               const float* __restrict__ shift, size_t nquads) {
    size_t i = (size_t)blockIdx.x * blockDim.x + threadIdx.x;
    if (i >= nquads) return;
    int c = (int)((i * 4) & 127);
    float4 v = ((const float4*)X)[i];
    float4 sc = *(const float4*)(scale + c);
    float4 sh = *(const float4*)(shift + c);
    v.x = v.x * sc.x + sh.x; v.x = (v.x >= 0.0f) ? v.x : NEG_SLOPE * v.x;
    v.y = v.y * sc.y + sh.y; v.y = (v.y >= 0.0f) ? v.y : NEG_SLOPE * v.y;
    v.z = v.z * sc.z + sh.z; v.z = (v.z >= 0.0f) ? v.z : NEG_SLOPE * v.z;
    v.w = v.w * sc.w + sh.w; v.w = (v.w >= 0.0f) ? v.w : NEG_SLOPE * v.w;
    ((float4*)X)[i] = v;
}

// ----------------------------------------------------------- pooling/heads
__global__ void pool_sum(const float* __restrict__ H, const int* __restrict__ gids,
                         float* __restrict__ gsum, float* __restrict__ gcnt, int nrows) {
    int gid  = blockIdx.x * blockDim.x + threadIdx.x;
    int i    = gid >> 5;
    int lane = gid & 31;
    if (i >= nrows) return;
    int g = gids[i];
    float4 v = ((const float4*)(H + (size_t)i * 128))[lane];
    float* out = gsum + (size_t)g * 128 + lane * 4;
    unsafeAtomicAdd(out + 0, v.x);
    unsafeAtomicAdd(out + 1, v.y);
    unsafeAtomicAdd(out + 2, v.z);
    unsafeAtomicAdd(out + 3, v.w);
    if (lane == 0) unsafeAtomicAdd(&gcnt[g], 1.0f);
}

// block handles 32 rows; rows + the 4KB weight staged through LDS (coalesced)
__global__ __launch_bounds__(256) void node_head(const float* __restrict__ Hn,
                                                 const float* __restrict__ Wnc,
                                                 const float* __restrict__ bnc,
                                                 float* __restrict__ out, int nrows) {
    __shared__ float sh[32 * 128];
    __shared__ float sw[128 * NC];
    int tid = threadIdx.x;
    int rbase = blockIdx.x * 32;
    for (int idx = tid; idx < 32 * 128; idx += 256) {
        int r = rbase + (idx >> 7);
        sh[idx] = (r < nrows) ? Hn[(size_t)r * 128 + (idx & 127)] : 0.0f;
    }
    for (int idx = tid; idx < 128 * NC; idx += 256) sw[idx] = Wnc[idx];
    __syncthreads();
    int lr = tid >> 3, c = tid & 7;
    int row = rbase + lr;
    float acc = bnc[c];
#pragma unroll 8
    for (int k = 0; k < 128; ++k) acc += sh[lr * 128 + k] * sw[k * NC + c];
    if (row < nrows) out[(size_t)row * NC + c] = acc;
}

__global__ void graph_head(const float* __restrict__ gsum, const float* __restrict__ gcnt,
                           const float* __restrict__ Wgc, const float* __restrict__ bgc,
                           float* __restrict__ out) {
    int t = threadIdx.x;  // 256 threads, need 64*4
    if (t >= NGRAPH * GC) return;
    int g = t >> 2, c = t & 3;
    float inv = 1.0f / fmaxf(gcnt[g], 1.0f);
    float acc = bgc[c];
    for (int k = 0; k < 128; ++k) acc += gsum[g * 128 + k] * inv * Wgc[k * GC + c];
    out[t] = acc;
}

// ---------------------------------------------------------------- launcher

static inline unsigned cdiv(size_t a, unsigned b) { return (unsigned)((a + b - 1) / b); }

extern "C" void kernel_launch(void* const* d_in, const int* in_sizes, int n_in,
                              void* d_out, int out_size, void* d_ws, size_t ws_size,
                              hipStream_t stream) {
    const float* node_feat = (const float*)d_in[0];
    const float* nodetype  = (const float*)d_in[1];
    const float* W1    = (const float*)d_in[3];
    const float* b1    = (const float*)d_in[4];
    const float* g1    = (const float*)d_in[5];
    const float* be1   = (const float*)d_in[6];
    const float* W2    = (const float*)d_in[7];
    const float* b2    = (const float*)d_in[8];
    const float* g2    = (const float*)d_in[9];
    const float* be2   = (const float*)d_in[10];
    const float* Wn1   = (const float*)d_in[13];
    const float* b_nn1 = (const float*)d_in[14];
    const float* Wn2   = (const float*)d_in[15];
    const float* b_nn2 = (const float*)d_in[16];
    const float* Wnc   = (const float*)d_in[17];
    const float* b_nc  = (const float*)d_in[18];
    const float* Wg1   = (const float*)d_in[19];
    const float* b_g1  = (const float*)d_in[20];
    const float* Wgc   = (const float*)d_in[21];
    const float* b_gc  = (const float*)d_in[22];
    const int* src  = (const int*)d_in[23];
    const int* dst  = (const int*)d_in[24];
    const int* gids = (const int*)d_in[25];

    const int N = in_sizes[1];
    const int E = in_sizes[23];
    const size_t NH  = (size_t)N * 128;
    const size_t PAD = 128 * 128;           // 128 spare rows per big buffer
    const size_t NHP = NH + PAD;

    // workspace carve-up
    float* p = (float*)d_ws;
    float* sOut = p;  p += N;               // out-degree -> rsqrt norm
    float* sIn  = p;  p += N;               // in-degree  -> rsqrt norm
    float* aggC = p;  p += N;               // nodetype column aggregation
    float* csum = p;  p += 128;
    float* csq  = p;  p += 128;
    float* bnsc = p;  p += 128;
    float* bnsh = p;  p += 128;
    float* gsum = p;  p += NGRAPH * 128;
    float* gcnt = p;  p += NGRAPH;
    float* A = p;     p += NHP;             // aggregation buffer (+padding)
    float* B = p;     p += NHP;
    float* C = p;     p += NHP;

    float* out_node  = (float*)d_out;              // [N, 8]
    float* out_graph = out_node + (size_t)N * NC;  // [64, 4]

    const unsigned TB = 256;
    unsigned gbN    = cdiv((size_t)N, TB);
    unsigned gbNH   = cdiv(NH, TB);
    unsigned gbNH4  = cdiv(NH / 4, TB);
    unsigned gbE    = cdiv((size_t)E, TB);
    unsigned gbE32  = cdiv((size_t)E * 32, TB);
    unsigned gbN32  = cdiv((size_t)N * 32, TB);
    unsigned gbGemm = cdiv((size_t)N, 128);
    unsigned gbStat = cdiv((size_t)N, 256);

    // degrees -> rsqrt norms
    zero_f32<<<gbN, TB, 0, stream>>>(sOut, (size_t)N);
    zero_f32<<<gbN, TB, 0, stream>>>(sIn, (size_t)N);
    degree_accum<<<gbE, TB, 0, stream>>>(src, dst, sOut, sIn, E);
    inv_sqrt_deg<<<gbN, TB, 0, stream>>>(sOut, N);
    inv_sqrt_deg<<<gbN, TB, 0, stream>>>(sIn, N);

    // layer 1: h1 = lrelu(bn(gconv(x, W1, b1)))
    zero_f32<<<gbNH, TB, 0, stream>>>(A, NH);
    scatter_add_128<<<gbE32, TB, 0, stream>>>(node_feat, sOut, src, dst, A, E);
    gemm128_wmma<false, false><<<gbGemm, TB, 0, stream>>>(A, W1, b1, sIn, nullptr, nullptr, B, N);
    zero_f32<<<1, 128, 0, stream>>>(csum, 128);
    zero_f32<<<1, 128, 0, stream>>>(csq, 128);
    col_stats<<<gbStat, 128, 0, stream>>>(B, csum, csq, N);
    bn_finalize<<<1, 128, 0, stream>>>(csum, csq, g1, be1, bnsc, bnsh, N);
    bn_apply_lrelu<<<gbNH4, TB, 0, stream>>>(B, bnsc, bnsh, NH / 4);

    // layer 2: h = lrelu(bn(gconv(h1, W2, b2)))   -> C
    zero_f32<<<gbNH, TB, 0, stream>>>(A, NH);
    scatter_add_128<<<gbE32, TB, 0, stream>>>(B, sOut, src, dst, A, E);
    gemm128_wmma<false, false><<<gbGemm, TB, 0, stream>>>(A, W2, b2, sIn, nullptr, nullptr, C, N);
    zero_f32<<<1, 128, 0, stream>>>(csum, 128);
    zero_f32<<<1, 128, 0, stream>>>(csq, 128);
    col_stats<<<gbStat, 128, 0, stream>>>(C, csum, csq, N);
    bn_finalize<<<1, 128, 0, stream>>>(csum, csq, g2, be2, bnsc, bnsh, N);
    bn_apply_lrelu<<<gbNH4, TB, 0, stream>>>(C, bnsc, bnsh, NH / 4);

    // agg3 = normalized aggregation of h (reused for both Wn1 layer and hg layer)
    zero_f32<<<gbNH, TB, 0, stream>>>(A, NH);
    scatter_add_128<<<gbE32, TB, 0, stream>>>(C, sOut, src, dst, A, E);

    // hn1 = lrelu(gconv(h, Wn1, b_nn1)) -> B
    gemm128_wmma<true, false><<<gbGemm, TB, 0, stream>>>(A, Wn1, b_nn1, sIn, nullptr, nullptr, B, N);

    // hn2 = lrelu(gconv(hn1, Wn2, b_nn2)) -> B   (agg in C; h no longer needed)
    zero_f32<<<gbNH, TB, 0, stream>>>(C, NH);
    scatter_add_128<<<gbE32, TB, 0, stream>>>(B, sOut, src, dst, C, E);
    gemm128_wmma<true, false><<<gbGemm, TB, 0, stream>>>(C, Wn2, b_nn2, sIn, nullptr, nullptr, B, N);

    // node head: out_node = hn2 @ Wnc + b_nc
    node_head<<<cdiv((size_t)N, 32), TB, 0, stream>>>(B, Wnc, b_nc, out_node, N);

    // hg layer: concat([h, nodetype]) handled as agg3 @ Wg1[0:128] + aggC * Wg1[128]
    zero_f32<<<gbN, TB, 0, stream>>>(aggC, (size_t)N);
    scatter_scalar<<<gbE, TB, 0, stream>>>(nodetype, sOut, src, dst, aggC, E);
    scale_vec<<<gbN, TB, 0, stream>>>(aggC, sIn, N);
    gemm128_wmma<true, true><<<gbGemm, TB, 0, stream>>>(A, Wg1, b_g1, sIn, aggC, Wg1 + 128 * 128, C, N);

    // mean pooling per graph + graph head
    zero_f32<<<cdiv((size_t)NGRAPH * 128, TB), TB, 0, stream>>>(gsum, (size_t)NGRAPH * 128);
    zero_f32<<<1, NGRAPH, 0, stream>>>(gcnt, (size_t)NGRAPH);
    pool_sum<<<gbN32, TB, 0, stream>>>(C, gids, gsum, gcnt, N);
    graph_head<<<1, TB, 0, stream>>>(gsum, gcnt, Wgc, b_gc, out_graph);
}
